// OptimizedLSSBasedTPVGeneratorV2_19361712570728
// MI455X (gfx1250) — matrix operand order repaired and checked by reading, not compiled, not run on hardware
//
#include <hip/hip_runtime.h>
#include <hip/hip_bf16.h>

typedef __attribute__((ext_vector_type(16))) _Float16 v16h;
typedef __attribute__((ext_vector_type(8)))  float    v8f;

#define NN    2
#define CC    128
#define HH    64
#define WW    176
#define HW    11264          // 64*176
#define DD    80
#define OO    208            // CC + DD
#define NPIX  22528          // NN*HW
#define TPVY  200
#define TPVX  704
#define TPVZ  32
#define PXY   140800         // 200*704
#define PXZ   22528          // 704*32
#define PYZ   6400           // 200*32

// ---------------------------------------------------------------------------
// Kernel A: depthnet GEMM via WMMA f16->f32.
// Per camera n: out[o, hw] = sum_c W[o,c] * feat[c, hw] + bias[o]
// One wave handles a 16-wide column strip and all 13 row tiles (M=208).
// ---------------------------------------------------------------------------
__global__ __launch_bounds__(128) void gemm_depthnet(
    const float* __restrict__ feat, const float* __restrict__ Wm,
    const float* __restrict__ bias, float* __restrict__ out) {
  const int n    = blockIdx.y;
  const int wave = threadIdx.x >> 5;
  const int lane = threadIdx.x & 31;
  const int col0 = (blockIdx.x * 4 + wave) * 16;
  const int nh   = lane & 15;          // row/col within tile for A/B/C layouts
  const bool hi  = lane >= 16;
  const int col  = col0 + nh;

  const float* fbase = feat + (size_t)n * CC * HW;

  v8f acc[13] = {};

  #pragma unroll
  for (int kb = 0; kb < CC; kb += 32) {
    // B operand: 32x16 f16. lanes 0-15: K=kb..kb+15, lanes 16-31: K=kb+16..kb+31
    v16h bt;
    const float* fp = fbase + (size_t)(kb + (hi ? 16 : 0)) * HW + col;
    #pragma unroll
    for (int j = 0; j < 16; ++j) bt[j] = (_Float16)fp[(size_t)j * HW];

    #pragma unroll
    for (int mt = 0; mt < 13; ++mt) {
      // A operand: 16x32 f16. lane row = nh; lanes<16: K=0..15, lanes>=16: K=16..31
      const float* wp = Wm + (size_t)(mt * 16 + nh) * CC + kb + (hi ? 16 : 0);
      v16h at;
      #pragma unroll
      for (int i = 0; i < 8; ++i) {
        at[i]     = (_Float16)wp[i];
        at[8 + i] = (_Float16)wp[8 + i];
      }
      acc[mt] = __builtin_amdgcn_wmma_f32_16x16x32_f16(
          false, at, false, bt, (short)0, acc[mt], false, false);
    }
  }

  float* obase = out + (size_t)n * OO * HW;
  const int rb = hi ? 8 : 0;
  #pragma unroll
  for (int mt = 0; mt < 13; ++mt) {
    #pragma unroll
    for (int j = 0; j < 8; ++j) {
      const int o = mt * 16 + rb + j;   // C/D layout: VGPR j -> row j (+8 hi half)
      obase[(size_t)o * HW + col] = acc[mt][j] + bias[o];
    }
  }
}

// ---------------------------------------------------------------------------
// Kernel B: per-pixel softmax over D=80 depth bins (+threshold) and feature
// transpose to pixel-major so the scatter stage reads contiguously.
// ---------------------------------------------------------------------------
__global__ void softmax_transpose(const float* __restrict__ out,
                                  float* __restrict__ probT,
                                  float* __restrict__ featT) {
  const int pix = blockIdx.x * blockDim.x + threadIdx.x;
  if (pix >= NPIX) return;
  const int n = pix / HW;
  const int hw = pix - n * HW;
  const float* p = out + (size_t)n * OO * HW + hw;

  float v[DD];
  float mx = -3.0e38f;
  #pragma unroll
  for (int d = 0; d < DD; ++d) {
    v[d] = p[(size_t)d * HW];
    mx = fmaxf(mx, v[d]);
  }
  float s = 0.f;
  #pragma unroll
  for (int d = 0; d < DD; ++d) { v[d] = __expf(v[d] - mx); s += v[d]; }
  const float inv = 1.0f / s;

  float* pt = probT + (size_t)pix * DD;
  #pragma unroll
  for (int d = 0; d < DD; ++d) {
    const float pr = v[d] * inv;
    pt[d] = (pr > 0.0001f) ? pr : 0.f;     // wprob (thresholded)
  }

  const float* fp = p + (size_t)DD * HW;
  float* ft = featT + (size_t)pix * CC;
  #pragma unroll 8
  for (int c = 0; c < CC; ++c) ft[c] = fp[(size_t)c * HW];
}

// ---------------------------------------------------------------------------
// Kernel C: geometry + run-length-merged scatter-add into cell-major planes.
// One 128-thread block per pixel; threads 0..79 compute per-depth voxel ids,
// 3 threads merge consecutive-equal voxel runs per plane (big atomic saver for
// the coarse xz/yz planes), then all 128 threads issue coalesced atomics.
// ---------------------------------------------------------------------------
__global__ __launch_bounds__(128) void scatter_tpv(
    const float* __restrict__ probT, const float* __restrict__ featT,
    const float* __restrict__ intr,  const float* __restrict__ extr,
    float* __restrict__ axy, float* __restrict__ axz, float* __restrict__ ayz) {
  __shared__ float sp[DD];
  __shared__ int   sixy[DD], sixz[DD], siyz[DD];
  __shared__ int   midx[3][DD];
  __shared__ float mwgt[3][DD];
  __shared__ int   mcnt[3];

  const int pix = blockIdx.x;
  const int n  = pix / HW;
  const int hw = pix - n * HW;
  const int h  = hw / WW;
  const int w  = hw - h * WW;
  const int t  = threadIdx.x;

  if (t < DD) {
    // analytic 3x3 inverse of K
    const float* K = intr + n * 9;
    const float a00=K[0],a01=K[1],a02=K[2],a10=K[3],a11=K[4],a12=K[5],
                a20=K[6],a21=K[7],a22=K[8];
    const float det = a00*(a11*a22-a12*a21) - a01*(a10*a22-a12*a20)
                    + a02*(a10*a21-a11*a20);
    const float id = 1.0f / det;
    const float i00=(a11*a22-a12*a21)*id, i01=(a02*a21-a01*a22)*id, i02=(a01*a12-a02*a11)*id;
    const float i10=(a12*a20-a10*a22)*id, i11=(a00*a22-a02*a20)*id, i12=(a02*a10-a00*a12)*id;
    const float i20=(a10*a21-a11*a20)*id, i21=(a01*a20-a00*a21)*id, i22=(a00*a11-a01*a10)*id;

    const float u = (float)w, vv = (float)h;
    const float rx = i00*u + i01*vv + i02;
    const float ry = i10*u + i11*vv + i12;
    const float rz = i20*u + i21*vv + i22;

    const float* E = extr + n * 16;
    const float dsv = 2.0f + (48.0f / 79.0f) * (float)t;   // linspace(2,50,80)
    const float cx = rx * dsv, cy = ry * dsv, cz = rz * dsv;
    const float wx = E[0]*cx + E[1]*cy + E[2] *cz + E[3];
    const float wy = E[4]*cx + E[5]*cy + E[6] *cz + E[7];
    const float wz = E[8]*cx + E[9]*cy + E[10]*cz + E[11];

    int vx = (int)((wx + 54.0f) * (1.0f / 0.54f));   // trunc like astype(int32)
    int vy = (int)((wy + 54.0f) * (1.0f / 0.54f));
    int vz = (int)((wz +  5.0f) * (1.0f / 0.25f));
    vx = min(max(vx, 0), TPVX - 1);
    vy = min(max(vy, 0), TPVY - 1);
    vz = min(max(vz, 0), TPVZ - 1);

    sp[t]   = probT[(size_t)pix * DD + t];
    sixy[t] = vy * TPVX + vx;
    sixz[t] = vx * TPVZ + vz;
    siyz[t] = vy * TPVZ + vz;
  }
  __syncthreads();

  if (t < 3) {
    const int* src = (t == 0) ? sixy : (t == 1) ? sixz : siyz;
    int cnt = 0, cur = -1;
    float wsum = 0.f;
    for (int d = 0; d < DD; ++d) {
      const float pd = sp[d];
      if (pd <= 0.f) continue;
      if (src[d] == cur) {
        wsum += pd;
      } else {
        if (cur >= 0) { midx[t][cnt] = cur; mwgt[t][cnt] = wsum; ++cnt; }
        cur = src[d]; wsum = pd;
      }
    }
    if (cur >= 0) { midx[t][cnt] = cur; mwgt[t][cnt] = wsum; ++cnt; }
    mcnt[t] = cnt;
  }
  __syncthreads();

  const float fv = featT[(size_t)pix * CC + t];
  #pragma unroll
  for (int pl = 0; pl < 3; ++pl) {
    float* acc = (pl == 0) ? axy : (pl == 1) ? axz : ayz;
    const int cnt = mcnt[pl];
    for (int k = 0; k < cnt; ++k) {
      atomicAdd(acc + (size_t)midx[pl][k] * CC + t, fv * mwgt[pl][k]);
    }
  }
}

// ---------------------------------------------------------------------------
// Kernel D: cell-major [cell][128] -> channel-major [128][cell] via LDS tile.
// 64 cells x 128 channels per block; padded LDS avoids bank conflicts.
// ---------------------------------------------------------------------------
__global__ __launch_bounds__(256) void transpose_plane(
    const float* __restrict__ acc, float* __restrict__ dst, int P) {
  __shared__ float tile[CC * 65];
  const int cell0 = blockIdx.x * 64;
  const int t = threadIdx.x;
  #pragma unroll
  for (int it = 0; it < 32; ++it) {
    const int l = it * 256 + t;
    const int cl = l >> 7, ch = l & 127;
    tile[ch * 65 + cl] = acc[(size_t)(cell0 + cl) * CC + ch];
  }
  __syncthreads();
  #pragma unroll
  for (int it = 0; it < 32; ++it) {
    const int l = it * 256 + t;
    const int ch = l >> 6, cl = l & 63;
    dst[(size_t)ch * P + cell0 + cl] = tile[ch * 65 + cl];
  }
}

// ---------------------------------------------------------------------------
extern "C" void kernel_launch(void* const* d_in, const int* in_sizes, int n_in,
                              void* d_out, int out_size, void* d_ws, size_t ws_size,
                              hipStream_t stream) {
  const float* image_feat = (const float*)d_in[0];
  // d_in[1] (conf_map) is unused by the reference
  const float* intr = (const float*)d_in[2];
  const float* extr = (const float*)d_in[3];
  const float* Wm   = (const float*)d_in[4];
  const float* bias = (const float*)d_in[5];
  float* out = (float*)d_out;

  float* ws = (float*)d_ws;
  size_t off = 0;
  float* ws_out = ws + off; off += (size_t)NN * OO * HW;   // 4,685,824
  float* featT  = ws + off; off += (size_t)NPIX * CC;      // 2,883,584
  float* probT  = ws + off; off += (size_t)NPIX * DD;      // 1,802,240
  float* acc_xy = ws + off; off += (size_t)PXY * CC;       // 18,022,400
  float* acc_xz = ws + off; off += (size_t)PXZ * CC;       // 2,883,584
  float* acc_yz = ws + off; off += (size_t)PYZ * CC;       //   819,200

  // zero only the accumulation planes (contiguous region)
  hipMemsetAsync(acc_xy, 0,
                 ((size_t)PXY + PXZ + PYZ) * CC * sizeof(float), stream);

  // A: WMMA GEMM. 704 column tiles / 4 waves per block = 176 blocks per camera
  gemm_depthnet<<<dim3(176, NN), 128, 0, stream>>>(image_feat, Wm, bias, ws_out);

  // B: softmax + threshold + feature transpose
  softmax_transpose<<<(NPIX + 255) / 256, 256, 0, stream>>>(ws_out, probT, featT);

  // C: geometry + merged scatter-add (one 128-thread block per pixel)
  scatter_tpv<<<NPIX, 128, 0, stream>>>(probT, featT, intr, extr,
                                        acc_xy, acc_xz, acc_yz);

  // D: transpose planes into (B, C, H', W') output layout
  transpose_plane<<<PXY / 64, 256, 0, stream>>>(acc_xy, out, PXY);
  transpose_plane<<<PXZ / 64, 256, 0, stream>>>(acc_xz, out + (size_t)CC * PXY, PXZ);
  transpose_plane<<<PYZ / 64, 256, 0, stream>>>(acc_yz,
                                                out + (size_t)CC * (PXY + PXZ), PYZ);
}
